// DualResidualVQ_33457795235844
// MI455X (gfx1250) — matrix operand order repaired and checked by reading, not compiled, not run on hardware
//
#include <hip/hip_runtime.h>
#include <hip/hip_bf16.h>

typedef __attribute__((ext_vector_type(16))) __bf16 v16bf;
typedef __attribute__((ext_vector_type(8)))  __bf16 v8bf;
typedef __attribute__((ext_vector_type(8)))  float  v8f;

#define L_LAYERS 4
#define KCODES   1024
#define D1C      256
#define D2C      256
#define CCH      512
#define BB       32
#define TT       2048
#define NTOK     (BB*TT)
#define MTILE    32
#define BLOCK    256

// ---------------- kernel 0a: zero workspace accumulators ----------------
__global__ void vq_zero(float* psum, unsigned* counts, float* part) {
    int i = blockIdx.x * blockDim.x + threadIdx.x;
    if (i < L_LAYERS * KCODES) { psum[i] = 0.f; counts[i] = 0u; }
    if (i < L_LAYERS * 4) part[i] = 0.f;
}

// ---------------- kernel 0b: L2-normalize codebook -> bf16 + |e|^2 ------
__global__ __launch_bounds__(BLOCK) void vq_norm_cb(const float* __restrict__ cb_kd,
                                                    __bf16* __restrict__ e1q,
                                                    float* __restrict__ colss) {
    int row  = blockIdx.x * (BLOCK / 32) + (threadIdx.x >> 5);   // 0 .. L*K-1
    int lane = threadIdx.x & 31;
    const float* src = cb_kd + (size_t)row * D1C;
    float v[8]; float ss = 0.f;
#pragma unroll
    for (int i = 0; i < 8; i++) { v[i] = src[lane + i * 32]; ss += v[i] * v[i]; }
#pragma unroll
    for (int m = 1; m < 32; m <<= 1) ss += __shfl_xor(ss, m, 32);
    float rs = rsqrtf(ss + 1e-12f);
    __bf16* dst = e1q + (size_t)row * D1C;
#pragma unroll
    for (int i = 0; i < 8; i++) dst[lane + i * 32] = (__bf16)(v[i] * rs);
    if (lane == 0) colss[row] = ss / (ss + 1e-12f);   // sum(e1q_norm^2)
}

// ---------------- kernel 1: fused residual VQ over all 4 layers ---------
__global__ __launch_bounds__(BLOCK) void vq_main(
    const float*  __restrict__ x,
    const float*  __restrict__ cb_kd,
    const float*  __restrict__ cb_gan,
    const __bf16* __restrict__ e1q,
    const float*  __restrict__ colss,
    float*        __restrict__ psum,
    unsigned*     __restrict__ counts,
    float*        __restrict__ part,
    float*        __restrict__ out)
{
    __shared__ float  resid[MTILE][CCH];      //  64 KB
    __shared__ float  dtile[MTILE][KCODES];   // 128 KB
    __shared__ __bf16 abuf[MTILE][D1C];       //  16 KB
    __shared__ float  rowss[MTILE];
    __shared__ float  row_lse[MTILE];
    __shared__ int    row_idx[L_LAYERS][MTILE];
    __shared__ float  sc[4];                  // mind, plogp, vq, gan partials

    const int tid  = threadIdx.x;
    const int wg   = blockIdx.x;
    const int n0   = wg * MTILE;
    const int b    = n0 / TT;
    const int t0   = n0 % TT;            // MTILE divides TT -> tile within one b
    const int wave = tid >> 5;
    const int lane = tid & 31;
    const int half = lane >> 4;          // 0/1
    const int l16  = lane & 15;
    const int l8   = lane & 7;
    const int q8   = lane >> 3;          // 0..3

    // ---- load x tile (32 consecutive t per wave -> fully coalesced) ----
    {
        int row   = tid & 31;
        int cbase = tid >> 5;            // 0..7
        const float* xb = x + (size_t)b * CCH * TT + t0 + row;
#pragma unroll 8
        for (int j = 0; j < 64; j++) {
            int c = cbase + 8 * j;
            resid[row][c] = xb[(size_t)c * TT];
        }
    }

    for (int l = 0; l < L_LAYERS; l++) {
        if (tid == 0) { sc[0] = 0.f; sc[1] = 0.f; sc[2] = 0.f; sc[3] = 0.f; }
        __syncthreads();

        // ---- step 1: l2-normalize z1 rows into bf16 A-buffer ----
        {
            int row = wave * 4 + q8;     // 8 waves x 4 rows
            float v[32]; float ss = 0.f;
#pragma unroll
            for (int i = 0; i < 32; i++) { v[i] = resid[row][l8 + i * 8]; ss += v[i] * v[i]; }
#pragma unroll
            for (int m = 1; m < 8; m <<= 1) ss += __shfl_xor(ss, m, 8);
            float rs = rsqrtf(ss + 1e-12f);
#pragma unroll
            for (int i = 0; i < 32; i++) abuf[row][l8 + i * 8] = (__bf16)(v[i] * rs);
            if (l8 == 0) rowss[row] = ss / (ss + 1e-12f);
        }
        __syncthreads();

        // ---- step 2: WMMA GEMM 32x1024 (wave owns 128 codes, 2 row tiles) ----
        {
            v8f acc[2][8];
#pragma unroll
            for (int rt = 0; rt < 2; rt++)
#pragma unroll
                for (int ct = 0; ct < 8; ct++)
#pragma unroll
                    for (int r = 0; r < 8; r++) acc[rt][ct][r] = 0.f;

            const __bf16* ebase = e1q + (size_t)l * KCODES * D1C;
#pragma unroll
            for (int ks = 0; ks < 8; ks++) {
                // batched B-fragment loads (one clause, then overlapped WMMAs)
                v16bf bf[8];
#pragma unroll
                for (int ct = 0; ct < 8; ct++) {
                    int code = wave * 128 + ct * 16 + l16;
                    bf[ct] = *(const v16bf*)(ebase + (size_t)code * D1C + ks * 32 + 16 * half);
                }
                // A fragments for the two row tiles (LDS)
                v16bf a[2];
#pragma unroll
                for (int rt = 0; rt < 2; rt++) {
                    const __bf16* ar = &abuf[rt * 16 + l16][ks * 32];
                    v8bf alo = *(const v8bf*)(ar + 16 * half);
                    v8bf ahi = *(const v8bf*)(ar + 16 * half + 8);
                    a[rt] = __builtin_shufflevector(alo, ahi,
                              0,1,2,3,4,5,6,7,8,9,10,11,12,13,14,15);
                }
#pragma unroll
                for (int ct = 0; ct < 8; ct++) {
                    acc[0][ct] = __builtin_amdgcn_wmma_f32_16x16x32_bf16(
                        false, a[0], false, bf[ct], (short)0, acc[0][ct], false, false);
                    acc[1][ct] = __builtin_amdgcn_wmma_f32_16x16x32_bf16(
                        false, a[1], false, bf[ct], (short)0, acc[1][ct], false, false);
                }
            }
            // d = |z1q|^2 - 2*dot + |e1q|^2 -> LDS d-tile
#pragma unroll
            for (int ct = 0; ct < 8; ct++) {
                int code = wave * 128 + ct * 16 + l16;
                float cs = colss[l * KCODES + code];
#pragma unroll
                for (int rt = 0; rt < 2; rt++)
#pragma unroll
                    for (int r = 0; r < 8; r++) {
                        int row = rt * 16 + r + 8 * half;
                        dtile[row][code] = rowss[row] - 2.f * acc[rt][ct][r] + cs;
                    }
            }
        }
        __syncthreads();

        // ---- step 3: per-row argmin + streaming log-softmax entropy ----
        {
            int row = wave * 4 + q8;     // 8 lanes scan one row
            float mind = 3.4e38f; int mi = 0;
            float m = -3.4e38f, Z = 0.f, S = 0.f;   // over s = -d
#pragma unroll 4
            for (int j = 0; j < 128; j++) {
                int k = l8 + j * 8;
                float d = dtile[row][k];
                if (d < mind) { mind = d; mi = k; }
                float s = -d;
                if (s > m) {
                    float e = __expf(m - s);
                    S = S * e + Z * e * (m - s);
                    Z = Z * e;
                    m = s;
                }
                float e2 = __expf(s - m);
                Z += e2;
                S += e2 * (s - m);
            }
#pragma unroll
            for (int msk = 1; msk < 8; msk <<= 1) {
                float mind2 = __shfl_xor(mind, msk, 8);
                int   mi2   = __shfl_xor(mi,   msk, 8);
                if (mind2 < mind || (mind2 == mind && mi2 < mi)) { mind = mind2; mi = mi2; }
                float m2 = __shfl_xor(m, msk, 8);
                float Z2 = __shfl_xor(Z, msk, 8);
                float S2 = __shfl_xor(S, msk, 8);
                float mm = fmaxf(m, m2);
                float e1 = __expf(m - mm), e2 = __expf(m2 - mm);
                S = S * e1 + Z * e1 * (m - mm) + S2 * e2 + Z2 * e2 * (m2 - mm);
                Z = Z * e1 + Z2 * e2;
                m = mm;
            }
            if (l8 == 0) {
                row_idx[l][row] = mi;
                row_lse[row]    = m + __logf(Z);        // logsumexp of -d
                float plogp = S / Z - __logf(Z);        // sum_k p*logp
                atomicAdd(&sc[0], mind);
                atomicAdd(&sc[1], plogp);
            }
        }
        __syncthreads();

        // ---- step 4: p-sums (codebook entropy), counts, indices out ----
        {
#pragma unroll
            for (int j = 0; j < 4; j++) {
                int k = tid + j * 256;
                float s = 0.f;
#pragma unroll
                for (int row = 0; row < MTILE; row++)
                    s += __expf(-dtile[row][k] - row_lse[row]);
                atomicAdd(&psum[l * KCODES + k], s);
            }
            if (tid < MTILE) {
                int mi = row_idx[l][tid];
                atomicAdd(&counts[l * KCODES + mi], 1u);
                out[(size_t)BB * CCH * TT + (size_t)(n0 + tid) * L_LAYERS + l] = (float)mi;
            }

            // ---- step 5: residual update + vq/gan loss partials ----
            int row = tid >> 3;          // 32 rows, 8 threads each
            int c0  = tid & 7;
            int mi  = row_idx[l][row];
            const float* qkd  = cb_kd  + ((size_t)l * KCODES + mi) * D1C;
            const float* qgan = cb_gan + ((size_t)l * KCODES + mi) * D2C;
            float vq = 0.f, gan = 0.f;
#pragma unroll 8
            for (int j = 0; j < 64; j++) {
                int c = c0 + 8 * j;
                float q = (c < D1C) ? qkd[c] : qgan[c - D1C];
                float diff = q - resid[row][c];
                vq += diff * diff;
                if (c >= D1C) gan += diff * diff;
                resid[row][c] -= q;
            }
#pragma unroll
            for (int msk = 1; msk < 32; msk <<= 1) {
                vq  += __shfl_xor(vq,  msk, 32);
                gan += __shfl_xor(gan, msk, 32);
            }
            if (lane == 0) { atomicAdd(&sc[2], vq); atomicAdd(&sc[3], gan); }
        }
        __syncthreads();
        if (tid == 0) {
            atomicAdd(&part[l * 4 + 0], sc[2]);   // sum (q - z)^2 over N*512
            atomicAdd(&part[l * 4 + 1], sc[3]);   // sum (z2 - egan)^2 over N*256
            atomicAdd(&part[l * 4 + 2], sc[0]);   // sum min d
            atomicAdd(&part[l * 4 + 3], sc[1]);   // sum_k p logp (per row summed)
        }
        __syncthreads();
    }

    // ---- final: quantized_out = sum_l codebook[idx_l], x-layout store ----
    {
        int row   = tid & 31;
        int cbase = tid >> 5;
        int mis[L_LAYERS];
#pragma unroll
        for (int l = 0; l < L_LAYERS; l++) mis[l] = row_idx[l][row];
        float* ob = out + (size_t)b * CCH * TT + t0 + row;
#pragma unroll 8
        for (int j = 0; j < 64; j++) {
            int c = cbase + 8 * j;
            float q = 0.f;
#pragma unroll
            for (int l = 0; l < L_LAYERS; l++) {
                int mi = mis[l];
                q += (c < D1C) ? cb_kd [((size_t)l * KCODES + mi) * D1C + c]
                               : cb_gan[((size_t)l * KCODES + mi) * D2C + (c - D1C)];
            }
            ob[(size_t)c * TT] = q;
        }
    }
}

// ---------------- kernel 2: finalize 6 combined scalars -----------------
__global__ __launch_bounds__(BLOCK) void vq_final(const float* __restrict__ psum,
                                                  const unsigned* __restrict__ counts,
                                                  const float* __restrict__ part,
                                                  float* __restrict__ out) {
    __shared__ float susage[L_LAYERS];
    __shared__ float scbent[L_LAYERS];
    int tid = threadIdx.x;
    if (tid < L_LAYERS) { susage[tid] = 0.f; scbent[tid] = 0.f; }
    __syncthreads();
    for (int l = 0; l < L_LAYERS; l++) {
        float u = 0.f, ce = 0.f;
        for (int k = tid; k < KCODES; k += BLOCK) {
            u += (counts[l * KCODES + k] > 0u) ? 1.f : 0.f;
            float avg = psum[l * KCODES + k] / (float)NTOK;
            ce += avg * __logf(avg + 1e-10f);
        }
#pragma unroll
        for (int m = 1; m < 32; m <<= 1) {
            u  += __shfl_xor(u,  m, 32);
            ce += __shfl_xor(ce, m, 32);
        }
        if ((tid & 31) == 0) { atomicAdd(&susage[l], u); atomicAdd(&scbent[l], ce); }
    }
    __syncthreads();
    if (tid == 0) {
        float o[6] = {0.f, 0.f, 0.f, 0.f, 0.f, 0.f};
        for (int l = 0; l < L_LAYERS; l++) {
            float vq         = part[l * 4 + 0] / ((float)NTOK * 512.f);
            float gan        = part[l * 4 + 1] / ((float)NTOK * 256.f);
            float mind       = part[l * 4 + 2] / (float)NTOK;
            float sample_ent = -part[l * 4 + 3] / (float)NTOK;
            float cb_ent     = -scbent[l];
            float usage      = susage[l] / (float)KCODES;
            o[0] += vq;
            o[1] += 0.25f * vq;
            o[2] += 0.1f * (sample_ent - cb_ent);
            o[3] += usage;
            o[4] += mind;
            o[5] += gan;
        }
        size_t base = (size_t)BB * CCH * TT + (size_t)NTOK * L_LAYERS;
        for (int i = 0; i < 6; i++) out[base + i] = o[i] / (float)L_LAYERS;
    }
}

// ---------------- launch ----------------
extern "C" void kernel_launch(void* const* d_in, const int* in_sizes, int n_in,
                              void* d_out, int out_size, void* d_ws, size_t ws_size,
                              hipStream_t stream) {
    const float* x      = (const float*)d_in[0];
    const float* cb_kd  = (const float*)d_in[1];
    const float* cb_gan = (const float*)d_in[2];
    float* out = (float*)d_out;

    char* ws = (char*)d_ws;
    __bf16*   e1q    = (__bf16*)ws;                                  // 2 MB
    float*    colss  = (float*)(ws + (size_t)L_LAYERS * KCODES * D1C * 2);
    float*    psum   = colss + L_LAYERS * KCODES;
    unsigned* counts = (unsigned*)(psum + L_LAYERS * KCODES);
    float*    part   = (float*)(counts + L_LAYERS * KCODES);

    vq_zero<<<(L_LAYERS * KCODES + BLOCK - 1) / BLOCK, BLOCK, 0, stream>>>(psum, counts, part);
    vq_norm_cb<<<(L_LAYERS * KCODES) / (BLOCK / 32), BLOCK, 0, stream>>>(cb_kd, e1q, colss);
    vq_main<<<NTOK / MTILE, BLOCK, 0, stream>>>(x, cb_kd, cb_gan, e1q, colss,
                                                psum, counts, part, out);
    vq_final<<<1, BLOCK, 0, stream>>>(psum, counts, part, out);
}
